// Attention_17918603559524
// MI455X (gfx1250) — matrix operand order repaired and checked by reading, not compiled, not run on hardware
//
#include <hip/hip_runtime.h>
#include <hip/hip_bf16.h>

// ---------------------------------------------------------------------------
// Types for CDNA5 WMMA bf16 path
// ---------------------------------------------------------------------------
typedef __attribute__((ext_vector_type(16))) __bf16 v16bf;
typedef __attribute__((ext_vector_type(8)))  __bf16 v8bf;
typedef __attribute__((ext_vector_type(8)))  float  v8f;

#define EMBED 1024
#define HEADS 16
#define HDIM  64
#define SEQ   2048
#define BATCH 2
#define ROWS  (BATCH * SEQ)        // 4096 token rows

__device__ __forceinline__ __bf16 f2bf(float f) {
    union { float f; unsigned u; } v; v.f = f;
    unsigned r = v.u + 0x7FFFu + ((v.u >> 16) & 1u);   // RNE
    unsigned short h = (unsigned short)(r >> 16);
    return __builtin_bit_cast(__bf16, h);
}

// A-fragment (16x32 bf16, MxK): lane L -> row m = L%16, half h = L/16.
// Elements: k = 16h + i (i=0..15) contiguous
__device__ __forceinline__ v16bf load_a_frag(const __bf16* rowptr, int kbase, int half) {
    return *(const v16bf*)(rowptr + kbase + 16 * half);
}

// B-fragment (32x16 bf16, KxN), Bc layout = per-column (n) contiguous K:
// lane L -> col n = L%16, half h = L/16, elements k = 16h + 0..15 contiguous.
__device__ __forceinline__ v16bf load_b_frag(const __bf16* colptr, int kbase, int half) {
    return *(const v16bf*)(colptr + kbase + 16 * half);
}

__device__ __forceinline__ v8f wmma_bf16(v16bf a, v16bf b, v8f c) {
    return __builtin_amdgcn_wmma_f32_16x16x32_bf16(
        /*neg_a=*/false, a, /*neg_b=*/false, b,
        /*c_mod=*/(short)0, c, /*reuse_a=*/false, /*reuse_b=*/false);
}

// ---------------------------------------------------------------------------
// fp32 -> bf16 conversion
// ---------------------------------------------------------------------------
__global__ void k_f32_to_bf16(const float* __restrict__ in, __bf16* __restrict__ out, int n) {
    int i = blockIdx.x * 256 + threadIdx.x;
    if (i < n) out[i] = f2bf(in[i]);
}

// W[h][e][d] (fp32) -> Wt[(h*64+d)][e] (bf16)  ("Bc" layout for projections)
__global__ void k_pack_w(const float* __restrict__ in, __bf16* __restrict__ out) {
    int i = blockIdx.x * 256 + threadIdx.x;        // 1024*1024 threads
    int n = i >> 10, e = i & 1023;
    int h = n >> 6,  d = n & 63;
    out[(size_t)n * EMBED + e] = f2bf(in[((size_t)h * EMBED + e) * HDIM + d]);
}

// ---------------------------------------------------------------------------
// Core 32x64 wave-tile, double-buffered (manually unrolled x2 so the two
// fragment buffers are named variables -> no dynamic register indexing).
// Fragments for step k+32 are in flight while the 8 WMMAs of step k issue.
// ---------------------------------------------------------------------------
#define GEMM_BODY(ACC0, ACC1)                                                  \
    v16bf a0A, a1A, bA0, bA1, bA2, bA3;                                        \
    v16bf a0B, a1B, bB0, bB1, bB2, bB3;                                        \
    a0A = load_a_frag(arow0, 0, half);                                         \
    a1A = load_a_frag(arow1, 0, half);                                         \
    bA0 = load_b_frag(bcol0, 0, half);                                         \
    bA1 = load_b_frag(bcol1, 0, half);                                         \
    bA2 = load_b_frag(bcol2, 0, half);                                         \
    bA3 = load_b_frag(bcol3, 0, half);                                         \
    for (int kk = 0; kk < EMBED; kk += 64) {                                   \
        /* loads for k = kk+32 (always in range: kk <= EMBED-64) */            \
        a0B = load_a_frag(arow0, kk + 32, half);                               \
        a1B = load_a_frag(arow1, kk + 32, half);                               \
        bB0 = load_b_frag(bcol0, kk + 32, half);                               \
        bB1 = load_b_frag(bcol1, kk + 32, half);                               \
        bB2 = load_b_frag(bcol2, kk + 32, half);                               \
        bB3 = load_b_frag(bcol3, kk + 32, half);                               \
        __builtin_prefetch(arow0 + kk + 160, 0, 3);                            \
        ACC0[0] = wmma_bf16(a0A, bA0, ACC0[0]);                                \
        ACC1[0] = wmma_bf16(a1A, bA0, ACC1[0]);                                \
        ACC0[1] = wmma_bf16(a0A, bA1, ACC0[1]);                                \
        ACC1[1] = wmma_bf16(a1A, bA1, ACC1[1]);                                \
        ACC0[2] = wmma_bf16(a0A, bA2, ACC0[2]);                                \
        ACC1[2] = wmma_bf16(a1A, bA2, ACC1[2]);                                \
        ACC0[3] = wmma_bf16(a0A, bA3, ACC0[3]);                                \
        ACC1[3] = wmma_bf16(a1A, bA3, ACC1[3]);                                \
        if (kk + 64 < EMBED) { /* loads for k = kk+64 */                       \
            a0A = load_a_frag(arow0, kk + 64, half);                           \
            a1A = load_a_frag(arow1, kk + 64, half);                           \
            bA0 = load_b_frag(bcol0, kk + 64, half);                           \
            bA1 = load_b_frag(bcol1, kk + 64, half);                           \
            bA2 = load_b_frag(bcol2, kk + 64, half);                           \
            bA3 = load_b_frag(bcol3, kk + 64, half);                           \
        }                                                                      \
        ACC0[0] = wmma_bf16(a0B, bB0, ACC0[0]);                                \
        ACC1[0] = wmma_bf16(a1B, bB0, ACC1[0]);                                \
        ACC0[1] = wmma_bf16(a0B, bB1, ACC0[1]);                                \
        ACC1[1] = wmma_bf16(a1B, bB1, ACC1[1]);                                \
        ACC0[2] = wmma_bf16(a0B, bB2, ACC0[2]);                                \
        ACC1[2] = wmma_bf16(a1B, bB2, ACC1[2]);                                \
        ACC0[3] = wmma_bf16(a0B, bB3, ACC0[3]);                                \
        ACC1[3] = wmma_bf16(a1B, bB3, ACC1[3]);                                \
    }

// ---------------------------------------------------------------------------
// QKV projection GEMM: [4096 x 1024] x [1024 x 1024] -> bf16
// Block = 128 thr (4 waves). Wave tile 32x64, workgroup tile 128x64.
// mode 0/1: store [b,h,s,d];  mode 2: store transposed [b,h,d,s]
// ---------------------------------------------------------------------------
__global__ void __launch_bounds__(128)
k_gemm_qkv(const __bf16* __restrict__ A, const __bf16* __restrict__ Bc,
           __bf16* __restrict__ out, int mode) {
    const int lane = threadIdx.x & 31;
    const int wave = threadIdx.x >> 5;
    const int half = lane >> 4;
    const int ln   = lane & 15;
    const int m0 = blockIdx.y * 128 + wave * 32;
    const int n0 = blockIdx.x * 64;

    const __bf16* arow0 = A + (size_t)(m0 + ln) * EMBED;
    const __bf16* arow1 = arow0 + (size_t)16 * EMBED;
    const __bf16* bcol0 = Bc + (size_t)(n0 +  0 + ln) * EMBED;
    const __bf16* bcol1 = Bc + (size_t)(n0 + 16 + ln) * EMBED;
    const __bf16* bcol2 = Bc + (size_t)(n0 + 32 + ln) * EMBED;
    const __bf16* bcol3 = Bc + (size_t)(n0 + 48 + ln) * EMBED;

    v8f zero8 = {0.f,0.f,0.f,0.f,0.f,0.f,0.f,0.f};
    v8f acc0[4], acc1[4];
#pragma unroll
    for (int t = 0; t < 4; ++t) { acc0[t] = zero8; acc1[t] = zero8; }

    GEMM_BODY(acc0, acc1)

    // C/D layout: lane -> n = n0+t*16+ln, VGPR j -> m = mg + j + 8*half
    auto store_group = [&](const v8f (&accp)[4], int mg) {
#pragma unroll
        for (int t = 0; t < 4; ++t) {
#pragma unroll
            for (int j = 0; j < 8; ++j) {
                int m = mg + j + 8 * half;       // = b*2048 + s
                int n = n0 + t * 16 + ln;        // = h*64 + d
                int b = m >> 11, s = m & 2047;
                int h = n >> 6,  d = n & 63;
                __bf16 v = f2bf(accp[t][j]);
                if (mode == 2)
                    out[((size_t)(b * HEADS + h) * HDIM + d) * SEQ + s] = v;  // V^T
                else
                    out[((size_t)(b * HEADS + h) * SEQ + s) * HDIM + d] = v;  // Q,K
            }
        }
    };
    store_group(acc0, m0);
    store_group(acc1, m0 + 16);
}

// ---------------------------------------------------------------------------
// Flash attention: workgroup = (b,h, 128-query tile), 4 waves x 32 rows.
// Online softmax, 64-key tiles. Each K/V fragment feeds 2 WMMAs (2 q-groups).
// ---------------------------------------------------------------------------
__global__ void __launch_bounds__(128)
k_attn(const __bf16* __restrict__ Q, const __bf16* __restrict__ K,
       const __bf16* __restrict__ Vt, __bf16* __restrict__ MH) {
    __shared__ __bf16 probs[4][32][64];          // per-wave 32x64 prob tile (16 KB)

    const int lane = threadIdx.x & 31;
    const int wave = threadIdx.x >> 5;
    const int half = lane >> 4;
    const int ln   = lane & 15;
    const int bh = blockIdx.y;                   // 0..31
    const int b  = bh >> 4, h = bh & 15;
    const int q0 = blockIdx.x * 128 + wave * 32;

    const __bf16* Qh = Q  + (size_t)bh * SEQ * HDIM;
    const __bf16* Kh = K  + (size_t)bh * SEQ * HDIM;
    const __bf16* Vh = Vt + (size_t)bh * HDIM * SEQ;

    v8f zero8 = {0.f,0.f,0.f,0.f,0.f,0.f,0.f,0.f};
    v8f o[2][4];
    float mrow[2][8], lrow[2][8];
#pragma unroll
    for (int g = 0; g < 2; ++g) {
#pragma unroll
        for (int t = 0; t < 4; ++t) o[g][t] = zero8;
#pragma unroll
        for (int j = 0; j < 8; ++j) { mrow[g][j] = -1e30f; lrow[g][j] = 0.f; }
    }

    // Loop-invariant Q A-fragments: 2 row groups x 2 k-chunks
    v16bf qa[2][2];
#pragma unroll
    for (int g = 0; g < 2; ++g) {
        const __bf16* qrow = Qh + (size_t)(q0 + g * 16 + ln) * HDIM;
        qa[g][0] = load_a_frag(qrow, 0,  half);
        qa[g][1] = load_a_frag(qrow, 32, half);
    }

    for (int kt = 0; kt < SEQ; kt += 64) {
        // ---- scores: 32x64 = Q(32x64) x K^T(64x64) ----
        v8f sc[2][4];
#pragma unroll
        for (int g = 0; g < 2; ++g)
#pragma unroll
            for (int t = 0; t < 4; ++t) sc[g][t] = zero8;
#pragma unroll
        for (int t = 0; t < 4; ++t) {
            const __bf16* kc = Kh + (size_t)(kt + t * 16 + ln) * HDIM;
            v16bf kb0 = load_b_frag(kc, 0,  half);
            v16bf kb1 = load_b_frag(kc, 32, half);
            sc[0][t] = wmma_bf16(qa[0][0], kb0, sc[0][t]);
            sc[0][t] = wmma_bf16(qa[0][1], kb1, sc[0][t]);
            sc[1][t] = wmma_bf16(qa[1][0], kb0, sc[1][t]);
            sc[1][t] = wmma_bf16(qa[1][1], kb1, sc[1][t]);
        }

        // ---- online softmax per row group ----
#pragma unroll
        for (int g = 0; g < 2; ++g) {
#pragma unroll
            for (int t = 0; t < 4; ++t)
#pragma unroll
                for (int j = 0; j < 8; ++j) sc[g][t][j] *= 0.125f;   // 1/sqrt(64)

            float mnew[8], alpha[8];
#pragma unroll
            for (int j = 0; j < 8; ++j) {
                float cur = fmaxf(fmaxf(sc[g][0][j], sc[g][1][j]),
                                  fmaxf(sc[g][2][j], sc[g][3][j]));
#pragma unroll
                for (int off = 1; off < 16; off <<= 1)
                    cur = fmaxf(cur, __shfl_xor(cur, off, 32));
                mnew[j]  = fmaxf(mrow[g][j], cur);
                alpha[j] = __expf(mrow[g][j] - mnew[j]);
                mrow[g][j] = mnew[j];
            }
            float rs[8];
#pragma unroll
            for (int j = 0; j < 8; ++j) rs[j] = 0.f;
#pragma unroll
            for (int t = 0; t < 4; ++t)
#pragma unroll
                for (int j = 0; j < 8; ++j) {
                    float p = __expf(sc[g][t][j] - mnew[j]);
                    sc[g][t][j] = p;
                    rs[j] += p;
                }
#pragma unroll
            for (int j = 0; j < 8; ++j) {
                float s = rs[j];
#pragma unroll
                for (int off = 1; off < 16; off <<= 1)
                    s += __shfl_xor(s, off, 32);
                lrow[g][j] = lrow[g][j] * alpha[j] + s;
#pragma unroll
                for (int t = 0; t < 4; ++t) o[g][t][j] *= alpha[j];
            }

            // ---- C-layout -> A-layout via LDS ----
#pragma unroll
            for (int t = 0; t < 4; ++t)
#pragma unroll
                for (int j = 0; j < 8; ++j)
                    probs[wave][g * 16 + j + 8 * half][t * 16 + ln] = f2bf(sc[g][t][j]);
        }
        __syncthreads();

        // ---- O += P(32x64) x V(64x64) ----
        const __bf16* prow0 = &probs[wave][ln][0];
        const __bf16* prow1 = &probs[wave][16 + ln][0];
#pragma unroll
        for (int c = 0; c < 2; ++c) {
            v16bf pa0 = load_a_frag(prow0, c * 32, half);
            v16bf pa1 = load_a_frag(prow1, c * 32, half);
#pragma unroll
            for (int t = 0; t < 4; ++t) {
                const __bf16* vc = Vh + (size_t)(t * 16 + ln) * SEQ + kt;
                v16bf vb = load_b_frag(vc, c * 32, half);
                o[0][t] = wmma_bf16(pa0, vb, o[0][t]);
                o[1][t] = wmma_bf16(pa1, vb, o[1][t]);
            }
        }
        __syncthreads();
    }

    // store multihead-concat layout: MH[(b*S+q)][h*64+d]  (bf16 for out-proj)
#pragma unroll
    for (int g = 0; g < 2; ++g)
#pragma unroll
        for (int t = 0; t < 4; ++t)
#pragma unroll
            for (int j = 0; j < 8; ++j) {
                int q = q0 + g * 16 + j + 8 * half;
                int d = t * 16 + ln;
                MH[(size_t)(b * SEQ + q) * EMBED + h * HDIM + d] =
                    f2bf(o[g][t][j] / lrow[g][j]);
            }
}

// ---------------------------------------------------------------------------
// Output projection: Y = MH @ Wo^T + b_o ; Bc[n][e] = Wo[n][e] (row-major Wo)
// Same 32x64 wave tile + double buffering; fp32 store with bias.
// ---------------------------------------------------------------------------
__global__ void __launch_bounds__(128)
k_out_proj(const __bf16* __restrict__ A, const __bf16* __restrict__ Bc,
           const float* __restrict__ bias, float* __restrict__ Y) {
    const int lane = threadIdx.x & 31;
    const int wave = threadIdx.x >> 5;
    const int half = lane >> 4;
    const int ln   = lane & 15;
    const int m0 = blockIdx.y * 128 + wave * 32;
    const int n0 = blockIdx.x * 64;

    const __bf16* arow0 = A + (size_t)(m0 + ln) * EMBED;
    const __bf16* arow1 = arow0 + (size_t)16 * EMBED;
    const __bf16* bcol0 = Bc + (size_t)(n0 +  0 + ln) * EMBED;
    const __bf16* bcol1 = Bc + (size_t)(n0 + 16 + ln) * EMBED;
    const __bf16* bcol2 = Bc + (size_t)(n0 + 32 + ln) * EMBED;
    const __bf16* bcol3 = Bc + (size_t)(n0 + 48 + ln) * EMBED;

    v8f zero8 = {0.f,0.f,0.f,0.f,0.f,0.f,0.f,0.f};
    v8f acc0[4], acc1[4];
#pragma unroll
    for (int t = 0; t < 4; ++t) { acc0[t] = zero8; acc1[t] = zero8; }

    GEMM_BODY(acc0, acc1)

    auto store_group = [&](const v8f (&accp)[4], int mg) {
#pragma unroll
        for (int t = 0; t < 4; ++t)
#pragma unroll
            for (int j = 0; j < 8; ++j) {
                int m = mg + j + 8 * half;
                int n = n0 + t * 16 + ln;
                Y[(size_t)m * EMBED + n] = accp[t][j] + bias[n];
            }
    };
    store_group(acc0, m0);
    store_group(acc1, m0 + 16);
}

// ---------------------------------------------------------------------------
// Launch
// ---------------------------------------------------------------------------
extern "C" void kernel_launch(void* const* d_in, const int* in_sizes, int n_in,
                              void* d_out, int out_size, void* d_ws, size_t ws_size,
                              hipStream_t stream) {
    const float* x  = (const float*)d_in[0];
    const float* Wq = (const float*)d_in[1];
    const float* Wk = (const float*)d_in[2];
    const float* Wv = (const float*)d_in[3];
    const float* Wo = (const float*)d_in[4];
    const float* bo = (const float*)d_in[5];
    float* out = (float*)d_out;

    char* ws = (char*)d_ws;
    __bf16* xb  = (__bf16*)(ws);                       //  8 MB  x bf16 [4096,1024]
    __bf16* wqb = (__bf16*)(ws + ( 8ull << 20));       //  2 MB  Wt_q
    __bf16* wkb = (__bf16*)(ws + (10ull << 20));       //  2 MB  Wt_k
    __bf16* wvb = (__bf16*)(ws + (12ull << 20));       //  2 MB  Wt_v
    __bf16* wob = (__bf16*)(ws + (14ull << 20));       //  2 MB  Wo bf16
    __bf16* Qb  = (__bf16*)(ws + (16ull << 20));       //  8 MB  Q [b,h,s,d]
    __bf16* Kb  = (__bf16*)(ws + (24ull << 20));       //  8 MB  K [b,h,s,d]
    __bf16* Vtb = (__bf16*)(ws + (32ull << 20));       //  8 MB  V^T [b,h,d,s]
    __bf16* MH  = (__bf16*)(ws + (40ull << 20));       //  8 MB  multihead concat

    k_f32_to_bf16<<<(ROWS * EMBED) / 256, 256, 0, stream>>>(x, xb, ROWS * EMBED);
    k_pack_w<<<(EMBED * EMBED) / 256, 256, 0, stream>>>(Wq, wqb);
    k_pack_w<<<(EMBED * EMBED) / 256, 256, 0, stream>>>(Wk, wkb);
    k_pack_w<<<(EMBED * EMBED) / 256, 256, 0, stream>>>(Wv, wvb);
    k_f32_to_bf16<<<(EMBED * EMBED) / 256, 256, 0, stream>>>(Wo, wob, EMBED * EMBED);

    dim3 gg(EMBED / 64, ROWS / 128);       // 16 x 32
    k_gemm_qkv<<<gg, 128, 0, stream>>>(xb, wqb, Qb, 0);
    k_gemm_qkv<<<gg, 128, 0, stream>>>(xb, wkb, Kb, 1);
    k_gemm_qkv<<<gg, 128, 0, stream>>>(xb, wvb, Vtb, 2);

    dim3 ga(SEQ / 128, BATCH * HEADS);     // 16 x 32
    k_attn<<<ga, 128, 0, stream>>>(Qb, Kb, Vtb, MH);

    k_out_proj<<<gg, 128, 0, stream>>>(MH, wob, bo, out);
}
